// HeteroTransportCell_23940147708207
// MI455X (gfx1250) — compile-verified, matrix operand order checked by reading, not run on hardware
//
#include <hip/hip_runtime.h>
#include <math.h>

// ---------------- problem constants (from reference) ----------------
#define CN1 20000
#define CN2 200000
#define CE11 40000
#define CE22 800000
#define CE12 100000

// ---------------- WMMA types & helpers ----------------
typedef __attribute__((ext_vector_type(16))) __bf16 v16bf;
typedef __attribute__((ext_vector_type(8)))  float  v8f;
typedef __attribute__((ext_vector_type(8)))  unsigned short u16x8;
typedef __attribute__((ext_vector_type(4)))  unsigned short u16x4;

#define LDS_SYNC() asm volatile("s_wait_dscnt 0" ::: "memory")

// fp32 -> bf16 via native conversion (lowers to v_cvt_pk_bf16_f32-class ops)
__device__ __forceinline__ unsigned short f2b(float f){
  union { __bf16 b; unsigned short s; } u;
  u.b = (__bf16)f;
  return u.s;
}
__device__ __forceinline__ unsigned pk2(float a, float b){
  return (unsigned)f2b(a) | ((unsigned)f2b(b) << 16);
}
__device__ __forceinline__ u16x4 pk4(float4 v){
  u16x4 r; r.x = f2b(v.x); r.y = f2b(v.y); r.z = f2b(v.z); r.w = f2b(v.w); return r;
}

__device__ __forceinline__ v16bf make_frag(u16x8 lo, u16x8 hi){
  union { u16x8 h[2]; v16bf v; } u; u.h[0] = lo; u.h[1] = hi; return u.v;
}

// A-matrix 16x32 bf16 fragment from bf16 row-major LDS tile [16][ldk]
// lane L: M = L%16 ; K = {0..7,16..23} (L<16) or {8..15,24..31} (L>=16)
__device__ __forceinline__ v16bf load_a_lds(const unsigned short* A, int ldk, int kt){
  int lane = threadIdx.x & 31;
  const unsigned short* p = A + (lane & 15) * ldk + kt * 32 + ((lane >> 4) << 3);
  u16x8 lo = *(const u16x8*)p;
  u16x8 hi = *(const u16x8*)(p + 16);
  return make_frag(lo, hi);
}

// B-matrix 32x16 bf16 fragment from pre-packed weights:
// packed[((nt*KT+kt)*32 + lane)*16 + i]
__device__ __forceinline__ v16bf load_b_pack(const unsigned short* Wp, int KT, int kt, int nt){
  int lane = threadIdx.x & 31;
  const unsigned short* p = Wp + ((size_t)((nt * KT + kt) * 32 + lane) << 4);
  u16x8 lo = *(const u16x8*)p;
  u16x8 hi = *(const u16x8*)(p + 8);
  return make_frag(lo, hi);
}

__device__ __forceinline__ v8f wmma_bf16(v16bf a, v16bf b, v8f c){
  return __builtin_amdgcn_wmma_f32_16x16x32_bf16(false, a, false, b, (short)0, c, false, false);
}

// C layout: lane L, vgpr g  <->  (M = g + 8*(L/16), N = L%16)
__device__ __forceinline__ void store_c_bf16(unsigned short* A, int ldk, int nt, v8f acc){
  int lane = threadIdx.x & 31; int N = lane & 15; int half = lane >> 4;
  #pragma unroll
  for (int g = 0; g < 8; ++g) A[(g + 8 * half) * ldk + nt * 16 + N] = f2b(acc[g]);
}
__device__ __forceinline__ void store_c_f32(float* U, int ldn, int nt, v8f acc){
  int lane = threadIdx.x & 31; int N = lane & 15; int half = lane >> 4;
  #pragma unroll
  for (int g = 0; g < 8; ++g) U[(g + 8 * half) * ldn + nt * 16 + N] = acc[g];
}

// one output n-tile: acc = act( A[16 x 32*KT] @ W^T + b )
__device__ __forceinline__ v8f gemm_nt(const unsigned short* A, int ldk,
                                       const unsigned short* Wp, int KT, int nt,
                                       const float* bias, bool relu){
  v8f acc = {0.f,0.f,0.f,0.f,0.f,0.f,0.f,0.f};
  for (int kt = 0; kt < KT; ++kt)
    acc = wmma_bf16(load_a_lds(A, ldk, kt), load_b_pack(Wp, KT, kt, nt), acc);
  float bv = bias ? bias[nt * 16 + (threadIdx.x & 15)] : 0.f;
  #pragma unroll
  for (int g = 0; g < 8; ++g){
    float v = acc[g] + bv;
    if (relu) v = fmaxf(v, 0.f);
    acc[g] = v;
  }
  return acc;
}

__device__ __forceinline__ float sigmoidf_(float x){ return 1.f / (1.f + expf(-x)); }
__device__ __forceinline__ float softplusf_(float x){ return (x > 20.f) ? x : log1pf(expf(x)); }

// monotone float<->uint encoding for atomicMax on floats
__device__ __forceinline__ unsigned fkey(float f){
  unsigned u = __float_as_uint(f);
  return (u & 0x80000000u) ? ~u : (u | 0x80000000u);
}
__device__ __forceinline__ float fdec(unsigned k){
  unsigned u = (k & 0x80000000u) ? (k & 0x7fffffffu) : ~k;
  return __uint_as_float(u);
}

// ---------------- weight packer: fp32 [dout][din] -> WMMA-B bf16 fragments ----------------
__global__ void pack_w_kernel(const float* __restrict__ W, unsigned short* __restrict__ dst,
                              int dout, int din, int KT){
  int NT = (dout + 15) >> 4;
  int total = NT * KT * 512;
  for (int idx = blockIdx.x * blockDim.x + threadIdx.x; idx < total; idx += gridDim.x * blockDim.x){
    int i    = idx & 15;
    int lane = (idx >> 4) & 31;
    int tk   = idx >> 9;
    int kt   = tk % KT;
    int nt   = tk / KT;
    int half = lane >> 4;
    int n = nt * 16 + (lane & 15);
    int k = kt * 32 + (i + ((i >= 8) ? 8 : 0)) + 8 * half;
    float v = (n < dout && k < din) ? W[(size_t)n * din + k] : 0.f;
    dst[idx] = f2b(v);
  }
}

// ---------------- edge message passing (StaticDynamicEdgeMP) ----------------
struct EdgeParams {
  const float *h_src, *h_dst, *xs_src, *xs_dst, *ea;
  const int *src, *dst;
  const unsigned short *embW1p, *embW2p, *gateW1p, *payW1p, *payW2p;
  const float *emb_b1, *emb_b2;
  const float *baseW, *baseB;
  const float *gate_b1, *gateW2, *gateB2;
  const float *pay_b1, *pay_b2;
  float* msg_out;
  int nE;
};

__global__ void __launch_bounds__(128) edge_mp_kernel(EdgeParams P){
  __shared__ __align__(16) unsigned short sA[4][16 * 128];
  __shared__ float sU[4][16 * 128];
  __shared__ float sS[4][16];
  __shared__ int   sDst[4][16];
  const int wave = threadIdx.x >> 5, lane = threadIdx.x & 31;
  const int M = lane & 15, half = lane >> 4;
  const int tile = blockIdx.x * 4 + wave;
  const int nT = (P.nE + 15) >> 4;
  if (tile >= nT) return;
  unsigned short* A = sA[wave];
  float* U = sU[wave];

  const int e0 = tile << 4;
  const bool valid = (e0 + M) < P.nE;
  const int eIdx = valid ? (e0 + M) : (P.nE - 1);
  const int s_n = P.src[eIdx];
  const int d_n = P.dst[eIdx];
  if (lane < 16) sDst[wave][M] = valid ? d_n : -1;

  // prefetch next tile's edge attrs (global_prefetch_b8)
  if (e0 + 16 + M < P.nE) __builtin_prefetch(P.ea + (size_t)(e0 + 16 + M) * 6, 0, 1);

  // ---- stage emb input [ea(6) | xs_src(8) | xs_dst(8) | 0-pad] -> A[16][32] bf16
  if (lane < 16){
    float t[32];
    const float* pe = P.ea + (size_t)eIdx * 6;
    #pragma unroll
    for (int c = 0; c < 6; ++c) t[c] = pe[c];
    float4 s0 = *(const float4*)(P.xs_src + (size_t)s_n * 8);
    float4 s1 = *(const float4*)(P.xs_src + (size_t)s_n * 8 + 4);
    float4 d0 = *(const float4*)(P.xs_dst + (size_t)d_n * 8);
    float4 d1 = *(const float4*)(P.xs_dst + (size_t)d_n * 8 + 4);
    t[6]  = s0.x; t[7]  = s0.y; t[8]  = s0.z; t[9]  = s0.w;
    t[10] = s1.x; t[11] = s1.y; t[12] = s1.z; t[13] = s1.w;
    t[14] = d0.x; t[15] = d0.y; t[16] = d0.z; t[17] = d0.w;
    t[18] = d1.x; t[19] = d1.y; t[20] = d1.z; t[21] = d1.w;
    #pragma unroll
    for (int c = 22; c < 32; ++c) t[c] = 0.f;
    unsigned* row32 = (unsigned*)(A + M * 32);
    #pragma unroll
    for (int j = 0; j < 16; ++j) row32[j] = pk2(t[2*j], t[2*j+1]);
  }
  LDS_SYNC();

  // ---- emb layer1: relu(in @ W1^T + b1), K=32 -> u1 bf16 in A[16][128]
  {
    v16bf a0 = load_a_lds(A, 32, 0);
    v8f outs[8];
    #pragma unroll
    for (int nt = 0; nt < 8; ++nt){
      v8f acc = {0.f,0.f,0.f,0.f,0.f,0.f,0.f,0.f};
      acc = wmma_bf16(a0, load_b_pack(P.embW1p, 1, 0, nt), acc);
      float bv = P.emb_b1[nt * 16 + M];
      #pragma unroll
      for (int g = 0; g < 8; ++g) acc[g] = fmaxf(acc[g] + bv, 0.f);
      outs[nt] = acc;
    }
    LDS_SYNC();
    #pragma unroll
    for (int nt = 0; nt < 8; ++nt) store_c_bf16(A, 128, nt, outs[nt]);
  }
  LDS_SYNC();

  // ---- emb layer2: u = relu(u1 @ W2^T + b2), K=128 -> fp32 U[16][128]
  #pragma unroll
  for (int nt = 0; nt < 8; ++nt)
    store_c_f32(U, 128, nt, gemm_nt(A, 128, P.embW2p, 4, nt, P.emb_b2, true));
  LDS_SYNC();

  // ---- base head: b_e = softplus(u . baseW + bb) — full wave, pairwise reduce
  float b_e = 0.f;
  {
    const float* Up = U + M * 128 + half * 64;
    const float* Wp = P.baseW + half * 64;
    float acc = 0.f;
    #pragma unroll 8
    for (int k = 0; k < 64; ++k) acc += Up[k] * Wp[k];
    acc += __shfl_xor(acc, 16);
    if (lane < 16) b_e = softplusf_(acc + P.baseB[0]);
  }

  // ---- stage gate input [hj | hi] bf16 into A[16][128]
  {
    const float* srcp = (half == 0) ? (P.h_src + (size_t)s_n * 64)
                                    : (P.h_dst + (size_t)d_n * 64);
    u16x4* row = (u16x4*)(A + M * 128 + half * 64);
    #pragma unroll
    for (int c = 0; c < 64; c += 4)
      row[c >> 2] = pk4(*(const float4*)(srcp + c));
  }
  LDS_SYNC();

  // ---- gate layer1 -> fp32 U (base head already consumed U)
  #pragma unroll
  for (int nt = 0; nt < 8; ++nt)
    store_c_f32(U, 128, nt, gemm_nt(A, 128, P.gateW1p, 4, nt, P.gate_b1, true));
  LDS_SYNC();

  // ---- gate head + combined scale — full wave, pairwise reduce
  {
    const float* Up = U + M * 128 + half * 64;
    const float* Wp = P.gateW2 + half * 64;
    float acc = 0.f;
    #pragma unroll 8
    for (int k = 0; k < 64; ++k) acc += Up[k] * Wp[k];
    acc += __shfl_xor(acc, 16);
    if (lane < 16) sS[wave][M] = valid ? (b_e * sigmoidf_(acc + P.gateB2[0])) : 0.f;
  }

  // ---- payload layer1: A cols 0..63 (= hj), K=64 -> bf16 back into A[16][128]
  {
    v16bf a0 = load_a_lds(A, 128, 0);
    v16bf a1 = load_a_lds(A, 128, 1);
    v8f outs[8];
    #pragma unroll
    for (int nt = 0; nt < 8; ++nt){
      v8f acc = {0.f,0.f,0.f,0.f,0.f,0.f,0.f,0.f};
      acc = wmma_bf16(a0, load_b_pack(P.payW1p, 2, 0, nt), acc);
      acc = wmma_bf16(a1, load_b_pack(P.payW1p, 2, 1, nt), acc);
      float bv = P.pay_b1[nt * 16 + M];
      #pragma unroll
      for (int g = 0; g < 8; ++g) acc[g] = fmaxf(acc[g] + bv, 0.f);
      outs[nt] = acc;
    }
    LDS_SYNC();
    #pragma unroll
    for (int nt = 0; nt < 8; ++nt) store_c_bf16(A, 128, nt, outs[nt]);
  }
  LDS_SYNC();

  // ---- payload layer2 + per-edge scale + scatter-add to dst
  #pragma unroll
  for (int nt = 0; nt < 4; ++nt){
    v8f acc = gemm_nt(A, 128, P.payW2p, 4, nt, P.pay_b2, false);
    #pragma unroll
    for (int g = 0; g < 8; ++g){
      int m  = g + 8 * half;
      int dn = sDst[wave][m];
      if (dn >= 0){
        float val = acc[g] * sS[wave][m];
        atomicAdd(P.msg_out + (size_t)dn * 64 + nt * 16 + M, val);
      }
    }
  }
}

// ---------------- dense [n,64] @ W[64,64]^T + b  (GAT xl / xr projections) ----------------
struct LinParams { const float* X; const unsigned short* Wp; const float* b; float* out; int n; };

__global__ void __launch_bounds__(128) lin64_kernel(LinParams P){
  __shared__ __align__(16) unsigned short sX[4][16 * 64];
  const int wave = threadIdx.x >> 5, lane = threadIdx.x & 31;
  const int M = lane & 15, half = lane >> 4;
  const int tile = blockIdx.x * 4 + wave;
  if (tile >= (P.n + 15) >> 4) return;
  const int base = tile << 4;
  const int node = (base + M < P.n) ? (base + M) : (P.n - 1);
  {
    const float* xp = P.X + (size_t)node * 64 + half * 32;
    u16x4* row = (u16x4*)(sX[wave] + M * 64 + half * 32);
    #pragma unroll
    for (int c = 0; c < 32; c += 4)
      row[c >> 2] = pk4(*(const float4*)(xp + c));
  }
  LDS_SYNC();
  #pragma unroll
  for (int nt = 0; nt < 4; ++nt){
    v8f acc = gemm_nt(sX[wave], 64, P.Wp, 2, nt, P.b, false);
    #pragma unroll
    for (int g = 0; g < 8; ++g){
      int nd = base + g + 8 * half;
      if (nd < P.n) P.out[(size_t)nd * 64 + nt * 16 + M] = acc[g];
    }
  }
}

// ---------------- GATv2 edge passes ----------------
__global__ void gat_logit_kernel(const float* __restrict__ xl, const float* __restrict__ xr,
                                 const float* __restrict__ att, const int* __restrict__ src,
                                 const int* __restrict__ dst, float* __restrict__ logit,
                                 unsigned* __restrict__ gmax, int nE){
  int tot = nE * 4;
  for (int idx = blockIdx.x * blockDim.x + threadIdx.x; idx < tot; idx += gridDim.x * blockDim.x){
    int e = idx >> 2, hh = idx & 3;
    int s = src[e], d = dst[e];
    const float* pl = xl + (size_t)s * 64 + hh * 16;
    const float* pr = xr + (size_t)d * 64 + hh * 16;
    const float* pa = att + hh * 16;
    float l = 0.f;
    #pragma unroll
    for (int c = 0; c < 16; ++c){
      float x = pl[c] + pr[c];
      x = (x > 0.f) ? x : 0.2f * x;       // leaky_relu 0.2
      l += x * pa[c];
    }
    logit[idx] = l;
    atomicMax(gmax + (size_t)d * 4 + hh, fkey(l));
  }
}

__global__ void gat_exp_kernel(float* __restrict__ logit, const unsigned* __restrict__ gmax,
                               float* __restrict__ gsum, const int* __restrict__ dst, int nE){
  int tot = nE * 4;
  for (int idx = blockIdx.x * blockDim.x + threadIdx.x; idx < tot; idx += gridDim.x * blockDim.x){
    int e = idx >> 2, hh = idx & 3;
    int d = dst[e];
    float m  = fdec(gmax[(size_t)d * 4 + hh]);
    float ee = expf(logit[idx] - m);
    logit[idx] = ee;
    atomicAdd(gsum + (size_t)d * 4 + hh, ee);
  }
}

__global__ void gat_scatter_kernel(const float* __restrict__ xl, const float* __restrict__ logit,
                                   const float* __restrict__ gsum, const int* __restrict__ src,
                                   const int* __restrict__ dst, float* __restrict__ m2, int nE){
  int tot = nE * 4;
  for (int idx = blockIdx.x * blockDim.x + threadIdx.x; idx < tot; idx += gridDim.x * blockDim.x){
    int e = idx >> 2, hh = idx & 3;
    int s = src[e], d = dst[e];
    float alpha = logit[idx] / gsum[(size_t)d * 4 + hh];
    const float* pl = xl + (size_t)s * 64 + hh * 16;
    float* po = m2 + (size_t)d * 64 + hh * 16;
    #pragma unroll
    for (int c = 0; c < 16; ++c) atomicAdd(po + c, alpha * pl[c]);
  }
}

// ---------------- GRU node update ----------------
struct GruParams {
  const float *h, *xd, *msg, *gbias;          // gbias may be null
  const float *dynW, *dynB;                   // [64][4], [64]
  const unsigned short *WihP, *WhhP;          // packed [192x128], [192x64]
  const float *bih, *bhh;                     // [192]
  float* out;
  int n;
};

__global__ void __launch_bounds__(128) gru_kernel(GruParams P){
  __shared__ __align__(16) unsigned short sX[4][16 * 128];
  __shared__ __align__(16) unsigned short sH[4][16 * 64];
  const int wave = threadIdx.x >> 5, lane = threadIdx.x & 31;
  const int M = lane & 15, half = lane >> 4;
  const int tile = blockIdx.x * 4 + wave;
  if (tile >= (P.n + 15) >> 4) return;
  unsigned short* X = sX[wave];
  unsigned short* H = sH[wave];
  const int base = tile << 4;
  const int node = (base + M < P.n) ? (base + M) : (P.n - 1);

  if (half == 0){
    // x[:,0:64] = dyn projection of xd ; also stage h (bf16) for gh GEMM
    float4 xdv = *(const float4*)(P.xd + (size_t)node * 4);
    const float* hw = P.h + (size_t)node * 64;
    u16x4* xrow = (u16x4*)(X + M * 128);
    u16x4* hrow = (u16x4*)(H + M * 64);
    #pragma unroll
    for (int c = 0; c < 64; c += 4){
      float4 o;
      {
        float4 w = *(const float4*)(P.dynW + (size_t)(c + 0) * 4);
        o.x = P.dynB[c+0] + xdv.x*w.x + xdv.y*w.y + xdv.z*w.z + xdv.w*w.w;
      }
      {
        float4 w = *(const float4*)(P.dynW + (size_t)(c + 1) * 4);
        o.y = P.dynB[c+1] + xdv.x*w.x + xdv.y*w.y + xdv.z*w.z + xdv.w*w.w;
      }
      {
        float4 w = *(const float4*)(P.dynW + (size_t)(c + 2) * 4);
        o.z = P.dynB[c+2] + xdv.x*w.x + xdv.y*w.y + xdv.z*w.z + xdv.w*w.w;
      }
      {
        float4 w = *(const float4*)(P.dynW + (size_t)(c + 3) * 4);
        o.w = P.dynB[c+3] + xdv.x*w.x + xdv.y*w.y + xdv.z*w.z + xdv.w*w.w;
      }
      xrow[c >> 2] = pk4(o);
      hrow[c >> 2] = pk4(*(const float4*)(hw + c));
    }
  } else {
    // x[:,64:128] = incoming messages (+ GAT bias for type-2 nodes)
    const float* mp = P.msg + (size_t)node * 64;
    u16x4* xrow = (u16x4*)(X + M * 128 + 64);
    #pragma unroll
    for (int c = 0; c < 64; c += 4){
      float4 v = *(const float4*)(mp + c);
      if (P.gbias){
        float4 gb = *(const float4*)(P.gbias + c);
        v.x += gb.x; v.y += gb.y; v.z += gb.z; v.w += gb.w;
      }
      xrow[c >> 2] = pk4(v);
    }
  }
  LDS_SYNC();

  v16bf ax[4], ah[2];
  #pragma unroll
  for (int kt = 0; kt < 4; ++kt) ax[kt] = load_a_lds(X, 128, kt);
  #pragma unroll
  for (int kt = 0; kt < 2; ++kt) ah[kt] = load_a_lds(H, 64, kt);

  #pragma unroll
  for (int t = 0; t < 4; ++t){
    v8f gr = {0.f,0.f,0.f,0.f,0.f,0.f,0.f,0.f}, gz = gr, gn = gr;
    v8f hr = gr, hz = gr, hn = gr;
    #pragma unroll
    for (int kt = 0; kt < 4; ++kt){
      gr = wmma_bf16(ax[kt], load_b_pack(P.WihP, 4, kt, t),     gr);
      gz = wmma_bf16(ax[kt], load_b_pack(P.WihP, 4, kt, 4 + t), gz);
      gn = wmma_bf16(ax[kt], load_b_pack(P.WihP, 4, kt, 8 + t), gn);
    }
    #pragma unroll
    for (int kt = 0; kt < 2; ++kt){
      hr = wmma_bf16(ah[kt], load_b_pack(P.WhhP, 2, kt, t),     hr);
      hz = wmma_bf16(ah[kt], load_b_pack(P.WhhP, 2, kt, 4 + t), hz);
      hn = wmma_bf16(ah[kt], load_b_pack(P.WhhP, 2, kt, 8 + t), hn);
    }
    float bir = P.bih[t * 16 + M], biz = P.bih[64 + t * 16 + M], bin = P.bih[128 + t * 16 + M];
    float bhr = P.bhh[t * 16 + M], bhz = P.bhh[64 + t * 16 + M], bhn = P.bhh[128 + t * 16 + M];
    #pragma unroll
    for (int g = 0; g < 8; ++g){
      int nd = base + g + 8 * half;
      float r  = sigmoidf_(gr[g] + bir + hr[g] + bhr);
      float z  = sigmoidf_(gz[g] + biz + hz[g] + bhz);
      float nn = tanhf(gn[g] + bin + r * (hn[g] + bhn));
      if (nd < P.n){
        float hp = P.h[(size_t)nd * 64 + t * 16 + M];
        P.out[(size_t)nd * 64 + t * 16 + M] = (1.f - z) * nn + z * hp;
      }
    }
  }
}

// ---------------- host launch ----------------
extern "C" void kernel_launch(void* const* d_in, const int* in_sizes, int n_in,
                              void* d_out, int out_size, void* d_ws, size_t ws_size,
                              hipStream_t stream){
  (void)in_sizes; (void)n_in; (void)out_size; (void)ws_size;

  const float* h1   = (const float*)d_in[0];
  const float* h2   = (const float*)d_in[1];
  const float* xd1  = (const float*)d_in[2];
  const float* xd2  = (const float*)d_in[3];
  const float* xs1  = (const float*)d_in[4];
  const float* xs2  = (const float*)d_in[5];
  const float* ea11 = (const float*)d_in[6];
  const float* ea22 = (const float*)d_in[7];
  // p11 (8..21), p22 (22..35), pgat (36..41), pnode1 (42..47), pnode2 (48..53)
  const int* ei11 = (const int*)d_in[54];
  const int* ei22 = (const int*)d_in[55];
  const int* ei12 = (const int*)d_in[56];

  char* wsBase = (char*)d_ws;
  size_t wsOff = 0;
  auto wsAlloc = [&](size_t bytes) -> void* {
    wsOff = (wsOff + 255) & ~(size_t)255;
    void* p = wsBase + wsOff;
    wsOff += bytes;
    return p;
  };

  auto packW = [&](const void* W, int dout, int din) -> unsigned short* {
    int KT = (din + 31) / 32, NT = (dout + 15) / 16;
    size_t elems = (size_t)NT * KT * 512;
    unsigned short* p = (unsigned short*)wsAlloc(elems * 2);
    int blocks = (int)((elems + 255) / 256);
    pack_w_kernel<<<blocks, 256, 0, stream>>>((const float*)W, p, dout, din, KT);
    return p;
  };

  // pack all GEMM weights into WMMA B-fragment order (bf16)
  unsigned short* p11_embW1 = packW(d_in[8],  128, 22);
  unsigned short* p11_embW2 = packW(d_in[10], 128, 128);
  unsigned short* p11_gatW1 = packW(d_in[14], 128, 128);
  unsigned short* p11_payW1 = packW(d_in[18], 128, 64);
  unsigned short* p11_payW2 = packW(d_in[20], 64, 128);
  unsigned short* p22_embW1 = packW(d_in[22], 128, 22);
  unsigned short* p22_embW2 = packW(d_in[24], 128, 128);
  unsigned short* p22_gatW1 = packW(d_in[28], 128, 128);
  unsigned short* p22_payW1 = packW(d_in[32], 128, 64);
  unsigned short* p22_payW2 = packW(d_in[34], 64, 128);
  unsigned short* gat_WlP   = packW(d_in[36], 64, 64);
  unsigned short* gat_WrP   = packW(d_in[38], 64, 64);
  unsigned short* n1_WihP   = packW(d_in[44], 192, 128);
  unsigned short* n1_WhhP   = packW(d_in[45], 192, 64);
  unsigned short* n2_WihP   = packW(d_in[50], 192, 128);
  unsigned short* n2_WhhP   = packW(d_in[51], 192, 64);

  float*    m1    = (float*)   wsAlloc((size_t)CN1 * 64 * 4);
  float*    m2    = (float*)   wsAlloc((size_t)CN2 * 64 * 4);
  float*    xl1   = (float*)   wsAlloc((size_t)CN1 * 64 * 4);
  float*    xr2   = (float*)   wsAlloc((size_t)CN2 * 64 * 4);
  float*    logit = (float*)   wsAlloc((size_t)CE12 * 4 * 4);
  unsigned* gmax  = (unsigned*)wsAlloc((size_t)CN2 * 4 * 4);
  float*    gsum  = (float*)   wsAlloc((size_t)CN2 * 4 * 4);

  hipMemsetAsync(m1,   0, (size_t)CN1 * 64 * 4, stream);
  hipMemsetAsync(m2,   0, (size_t)CN2 * 64 * 4, stream);
  hipMemsetAsync(gmax, 0, (size_t)CN2 * 4 * 4, stream);   // key(x) > 0 for any finite x
  hipMemsetAsync(gsum, 0, (size_t)CN2 * 4 * 4, stream);

  // GAT projections: xl1 = h1 @ Wl^T + bl ; xr2 = h2 @ Wr^T + br
  {
    LinParams L{h1, gat_WlP, (const float*)d_in[37], xl1, CN1};
    int tiles = (CN1 + 15) / 16;
    lin64_kernel<<<(tiles + 3) / 4, 128, 0, stream>>>(L);
  }
  {
    LinParams L{h2, gat_WrP, (const float*)d_in[39], xr2, CN2};
    int tiles = (CN2 + 15) / 16;
    lin64_kernel<<<(tiles + 3) / 4, 128, 0, stream>>>(L);
  }

  // edge type 1->1
  {
    EdgeParams P;
    P.h_src = h1; P.h_dst = h1; P.xs_src = xs1; P.xs_dst = xs1; P.ea = ea11;
    P.src = ei11; P.dst = ei11 + CE11;
    P.embW1p = p11_embW1; P.embW2p = p11_embW2; P.gateW1p = p11_gatW1;
    P.payW1p = p11_payW1; P.payW2p = p11_payW2;
    P.emb_b1 = (const float*)d_in[9];  P.emb_b2 = (const float*)d_in[11];
    P.baseW  = (const float*)d_in[12]; P.baseB  = (const float*)d_in[13];
    P.gate_b1= (const float*)d_in[15]; P.gateW2 = (const float*)d_in[16]; P.gateB2 = (const float*)d_in[17];
    P.pay_b1 = (const float*)d_in[19]; P.pay_b2 = (const float*)d_in[21];
    P.msg_out = m1; P.nE = CE11;
    int tiles = (CE11 + 15) / 16;
    edge_mp_kernel<<<(tiles + 3) / 4, 128, 0, stream>>>(P);
  }
  // edge type 2->2
  {
    EdgeParams P;
    P.h_src = h2; P.h_dst = h2; P.xs_src = xs2; P.xs_dst = xs2; P.ea = ea22;
    P.src = ei22; P.dst = ei22 + CE22;
    P.embW1p = p22_embW1; P.embW2p = p22_embW2; P.gateW1p = p22_gatW1;
    P.payW1p = p22_payW1; P.payW2p = p22_payW2;
    P.emb_b1 = (const float*)d_in[23]; P.emb_b2 = (const float*)d_in[25];
    P.baseW  = (const float*)d_in[26]; P.baseB  = (const float*)d_in[27];
    P.gate_b1= (const float*)d_in[29]; P.gateW2 = (const float*)d_in[30]; P.gateB2 = (const float*)d_in[31];
    P.pay_b1 = (const float*)d_in[33]; P.pay_b2 = (const float*)d_in[35];
    P.msg_out = m2; P.nE = CE22;
    int tiles = (CE22 + 15) / 16;
    edge_mp_kernel<<<(tiles + 3) / 4, 128, 0, stream>>>(P);
  }

  // GATv2 1->2: softmax over incoming edges per (dst, head), scatter xl*alpha into m2
  {
    const int* src12 = ei12;
    const int* dst12 = ei12 + CE12;
    const float* att = (const float*)d_in[40];
    int blocks = (CE12 * 4 + 255) / 256;
    gat_logit_kernel<<<blocks, 256, 0, stream>>>(xl1, xr2, att, src12, dst12, logit, gmax, CE12);
    gat_exp_kernel<<<blocks, 256, 0, stream>>>(logit, gmax, gsum, dst12, CE12);
    gat_scatter_kernel<<<blocks, 256, 0, stream>>>(xl1, logit, gsum, src12, dst12, m2, CE12);
  }

  float* out1 = (float*)d_out;
  float* out2 = out1 + (size_t)CN1 * 64;

  // GRU updates
  {
    GruParams G;
    G.h = h1; G.xd = xd1; G.msg = m1; G.gbias = nullptr;
    G.dynW = (const float*)d_in[42]; G.dynB = (const float*)d_in[43];
    G.WihP = n1_WihP; G.WhhP = n1_WhhP;
    G.bih = (const float*)d_in[46]; G.bhh = (const float*)d_in[47];
    G.out = out1; G.n = CN1;
    int tiles = (CN1 + 15) / 16;
    gru_kernel<<<(tiles + 3) / 4, 128, 0, stream>>>(G);
  }
  {
    GruParams G;
    G.h = h2; G.xd = xd2; G.msg = m2; G.gbias = (const float*)d_in[41]; // + GAT out bias
    G.dynW = (const float*)d_in[48]; G.dynB = (const float*)d_in[49];
    G.WihP = n2_WihP; G.WhhP = n2_WhhP;
    G.bih = (const float*)d_in[52]; G.bhh = (const float*)d_in[53];
    G.out = out2; G.n = CN2;
    int tiles = (CN2 + 15) / 16;
    gru_kernel<<<(tiles + 3) / 4, 128, 0, stream>>>(G);
  }
}
